// RGCNEncoder_73151882986060
// MI455X (gfx1250) — matrix operand order repaired and checked by reading, not compiled, not run on hardware
//
#include <hip/hip_runtime.h>

// ---------------------------------------------------------------------------
// RGCN encoder for MI455X (gfx1250), wave32, bf16 WMMA with f32 accumulation.
// B=128, R=6, N=256, D_IN=2048, D_H=512.
// ---------------------------------------------------------------------------

typedef __bf16 bf16;
typedef __attribute__((ext_vector_type(16))) __bf16 v16bf;
typedef __attribute__((ext_vector_type(8)))  __bf16 v8bf;
typedef __attribute__((ext_vector_type(8)))  float  v8f;

#define BATCH 128
#define NNODE 256
#define DIN   2048
#define DH    512
#define NREL  6

// ---------------------------------------------------------------------------
// Fragment loads (contiguous k order shared by A and B, 16-bit A 16x32 / B 32x16)
// A-frag: lane l -> row l%16; halves 0..15 = K base+(l/16)*16 + idx.
// ---------------------------------------------------------------------------
__device__ __forceinline__ v16bf load_a_bf16(const bf16* __restrict__ A, int ldA, int lane) {
    const int lo = lane & 15, hi = lane >> 4;
    const bf16* p = A + (size_t)lo * ldA + hi * 16;
    v8bf x0 = *(const v8bf*)p;
    v8bf x1 = *(const v8bf*)(p + 8);
    v16bf r;
#pragma unroll
    for (int i = 0; i < 8; ++i) { r[i] = x0[i]; r[8 + i] = x1[i]; }
    return r;
}

__device__ __forceinline__ v16bf load_a_f32(const float* __restrict__ A, int ldA, int lane) {
    const int lo = lane & 15, hi = lane >> 4;
    const float* p = A + (size_t)lo * ldA + hi * 16;
    v16bf r;
#pragma unroll
    for (int i = 0; i < 8; ++i) { r[i] = (__bf16)p[i]; r[8 + i] = (__bf16)p[i + 8]; }
    return r;
}

// B-frag: pre-packed 1KB blob per 32x16 tile; lane l reads 16 contiguous halfs.
__device__ __forceinline__ v16bf load_b_frag(const bf16* __restrict__ Bf, int lane) {
    return *(const v16bf*)(Bf + lane * 16);
}

__device__ __forceinline__ v8f wmma_bf16(v16bf a, v16bf b, v8f c) {
    return __builtin_amdgcn_wmma_f32_16x16x32_bf16(false, a, false, b, (short)0, c,
                                                   false, false);
}

// ---------------------------------------------------------------------------
// Pack a row-major f32 (K x Nc) weight matrix into B-fragment layout (bf16).
// Fragment (kTile,nTile) occupies 512 halfs; lane l holds col nTile*16+(l%16),
// K rows kTile*32 + (l/16)*16 + idx (idx = half index 0..15).
// blockIdx.z selects one of several stacked matrices (W_rel: 6).
// ---------------------------------------------------------------------------
__global__ void pack_wfrag(const float* __restrict__ src, bf16* __restrict__ dst,
                           int K, int Nc) {
    const size_t matElems = (size_t)K * Nc;
    const size_t base = (size_t)blockIdx.z * matElems;
    const size_t t = (size_t)blockIdx.x * 256 + threadIdx.x;
    if (t >= matElems) return;
    const int nTiles = Nc >> 4;
    const int fragIdx = (int)(t >> 9);
    const int lane = ((int)t >> 4) & 31;
    const int idx = (int)t & 15;
    const int kTile = fragIdx / nTiles;
    const int nTile = fragIdx % nTiles;
    const int n = (nTile << 4) + (lane & 15);
    const int k = (kTile << 5) + ((lane >> 4) << 4) + idx;
    dst[base + t] = (__bf16)src[base + (size_t)k * Nc + n];
}

// Repack activation X (B x N x DH, row-major bf16) into per-batch B-fragment
// layout for the E@X aggregation (K = N = 256 -> 8 kTiles, 32 nTiles).
__global__ void pack_xfrag(const bf16* __restrict__ X, bf16* __restrict__ Xf) {
    const size_t t = (size_t)blockIdx.x * 256 + threadIdx.x; // B * 131072 total
    const int b = (int)(t >> 17);
    const int u = (int)(t & 131071);
    const int fragIdx = u >> 9;
    const int lane = (u >> 4) & 31;
    const int idx = u & 15;
    const int kTile = fragIdx >> 5;      // nTiles = 32
    const int nTile = fragIdx & 31;
    const int n = (nTile << 4) + (lane & 15);
    const int k = (kTile << 5) + ((lane >> 4) << 4) + idx;
    Xf[t] = X[((size_t)b * NNODE + k) * DH + n];
}

// ---------------------------------------------------------------------------
// Embed: X0 = relu(attn_fts(32768x2048, f32) @ W_embed + b).  Out bf16.
// Grid: x = DH/64 = 8, y = (B*N)/128 = 256.  256 threads = 8 waves.
// ---------------------------------------------------------------------------
__global__ void __launch_bounds__(256) k_embed(const float* __restrict__ A,
                                               const bf16* __restrict__ Wf,
                                               const float* __restrict__ bias,
                                               bf16* __restrict__ out) {
    const int lane = threadIdx.x & 31;
    const int wave = threadIdx.x >> 5;
    const int nBlock = blockIdx.x * 64;
    const int nTileBase = nBlock >> 4;
    const int rowBase = blockIdx.y * 128 + wave * 16;
    const float* Arow = A + (size_t)rowBase * DIN;
    v8f acc[4] = {};
    for (int kb = 0; kb < DIN; kb += 32) {
        v16bf a = load_a_f32(Arow + kb, DIN, lane);
        const bf16* wk = Wf + ((size_t)(kb >> 5) * 32 + nTileBase) * 512;
#pragma unroll
        for (int f = 0; f < 4; ++f)
            acc[f] = wmma_bf16(a, load_b_frag(wk + f * 512, lane), acc[f]);
    }
    const int lo = lane & 15, hi = lane >> 4;
#pragma unroll
    for (int f = 0; f < 4; ++f) {
        const int col = nBlock + f * 16 + lo;
        const float bv = bias[col];
#pragma unroll
        for (int i = 0; i < 8; ++i) {
            float v = fmaxf(acc[f][i] + bv, 0.0f);
            out[(size_t)(rowBase + hi * 8 + i) * DH + col] = (__bf16)v;
        }
    }
}

// ---------------------------------------------------------------------------
// Aggregation: rel_fts[b,r] = rel_edges[b,r](256x256, f32) @ X[b](256x512).
// Grid: x = 8, y = 2, z = B*R = 768.
// ---------------------------------------------------------------------------
__global__ void __launch_bounds__(256) k_agg(const float* __restrict__ E,
                                             const bf16* __restrict__ Xfrag,
                                             bf16* __restrict__ relfts) {
    const int lane = threadIdx.x & 31;
    const int wave = threadIdx.x >> 5;
    const int z = blockIdx.z;               // b*NREL + r
    const int b = z / NREL;
    const int nBlock = blockIdx.x * 64;
    const int nTileBase = nBlock >> 4;
    const int rowBase = blockIdx.y * 128 + wave * 16;
    const float* Arow = E + (size_t)z * NNODE * NNODE + (size_t)rowBase * NNODE;
    const bf16* Wb = Xfrag + (size_t)b * (8 * 32 * 512);
    v8f acc[4] = {};
    for (int kb = 0; kb < NNODE; kb += 32) {
        v16bf a = load_a_f32(Arow + kb, NNODE, lane);
        const bf16* wk = Wb + ((size_t)(kb >> 5) * 32 + nTileBase) * 512;
#pragma unroll
        for (int f = 0; f < 4; ++f)
            acc[f] = wmma_bf16(a, load_b_frag(wk + f * 512, lane), acc[f]);
    }
    bf16* Crow = relfts + (size_t)z * NNODE * DH;
    const int lo = lane & 15, hi = lane >> 4;
#pragma unroll
    for (int f = 0; f < 4; ++f) {
        const int col = nBlock + f * 16 + lo;
#pragma unroll
        for (int i = 0; i < 8; ++i)
            Crow[(size_t)(rowBase + hi * 8 + i) * DH + col] = (__bf16)acc[f][i];
    }
}

// ---------------------------------------------------------------------------
// Combine: msgs[b] = relu( sum_r rel_fts[b,r] @ W_rel[r] + X[b] @ W_loop + b ).
// 7 accumulated K=512 GEMMs into the same f32 accumulators.
// Grid: x = 8, y = 2, z = B = 128.  outB (bf16) and/or outF (f32) nullable.
// ---------------------------------------------------------------------------
__global__ void __launch_bounds__(256) k_combine(const bf16* __restrict__ relfts,
                                                 const bf16* __restrict__ WrelF,
                                                 const bf16* __restrict__ Xrow,
                                                 const bf16* __restrict__ WloopF,
                                                 const float* __restrict__ bias,
                                                 bf16* __restrict__ outB,
                                                 float* __restrict__ outF) {
    const int lane = threadIdx.x & 31;
    const int wave = threadIdx.x >> 5;
    const int b = blockIdx.z;
    const int nBlock = blockIdx.x * 64;
    const int nTileBase = nBlock >> 4;
    const int rowBase = blockIdx.y * 128 + wave * 16;
    v8f acc[4] = {};
#pragma unroll 1
    for (int s = 0; s < NREL + 1; ++s) {
        const bf16* Arow;
        const bf16* Wf;
        if (s < NREL) {
            Arow = relfts + ((size_t)(b * NREL + s) * NNODE + rowBase) * DH;
            Wf = WrelF + (size_t)s * (16 * 32 * 512);
        } else {
            Arow = Xrow + ((size_t)b * NNODE + rowBase) * DH;
            Wf = WloopF;
        }
        for (int kb = 0; kb < DH; kb += 32) {
            v16bf a = load_a_bf16(Arow + kb, DH, lane);
            const bf16* wk = Wf + ((size_t)(kb >> 5) * 32 + nTileBase) * 512;
#pragma unroll
            for (int f = 0; f < 4; ++f)
                acc[f] = wmma_bf16(a, load_b_frag(wk + f * 512, lane), acc[f]);
        }
    }
    const int lo = lane & 15, hi = lane >> 4;
#pragma unroll
    for (int f = 0; f < 4; ++f) {
        const int col = nBlock + f * 16 + lo;
        const float bv = bias[col];
#pragma unroll
        for (int i = 0; i < 8; ++i) {
            float v = fmaxf(acc[f][i] + bv, 0.0f);
            const size_t off = ((size_t)b * NNODE + rowBase + hi * 8 + i) * DH + col;
            if (outB) outB[off] = (__bf16)v;
            if (outF) outF[off] = v;
        }
    }
}

// ---------------------------------------------------------------------------
extern "C" void kernel_launch(void* const* d_in, const int* in_sizes, int n_in,
                              void* d_out, int out_size, void* d_ws, size_t ws_size,
                              hipStream_t stream) {
    (void)in_sizes; (void)n_in; (void)out_size; (void)ws_size;
    const float* attn   = (const float*)d_in[0];
    const float* rel    = (const float*)d_in[1];
    const float* Wemb   = (const float*)d_in[2];
    const float* bemb   = (const float*)d_in[3];
    const float* Wrel0  = (const float*)d_in[4];
    const float* Wloop0 = (const float*)d_in[5];
    const float* b0     = (const float*)d_in[6];
    const float* Wrel1  = (const float*)d_in[7];
    const float* Wloop1 = (const float*)d_in[8];
    const float* b1     = (const float*)d_in[9];
    float* out = (float*)d_out;

    char* ws = (char*)d_ws;
    size_t off = 0;
    auto alloc = [&](size_t bytes) -> void* {
        void* p = ws + off;
        off += (bytes + 255) & ~(size_t)255;
        return p;
    };
    bf16* wembF   = (bf16*)alloc((size_t)DIN * DH * 2);
    bf16* wrel0F  = (bf16*)alloc((size_t)NREL * DH * DH * 2);
    bf16* wloop0F = (bf16*)alloc((size_t)DH * DH * 2);
    bf16* wrel1F  = (bf16*)alloc((size_t)NREL * DH * DH * 2);
    bf16* wloop1F = (bf16*)alloc((size_t)DH * DH * 2);
    bf16* Xrow0   = (bf16*)alloc((size_t)BATCH * NNODE * DH * 2);
    bf16* Xrow1   = (bf16*)alloc((size_t)BATCH * NNODE * DH * 2);
    bf16* Xfrag   = (bf16*)alloc((size_t)BATCH * NNODE * DH * 2);
    bf16* relfts  = (bf16*)alloc((size_t)BATCH * NREL * NNODE * DH * 2);

    // One-time weight packs into B-fragment layout.
    pack_wfrag<<<dim3((DIN * DH + 255) / 256, 1, 1), 256, 0, stream>>>(Wemb, wembF, DIN, DH);
    pack_wfrag<<<dim3((DH * DH + 255) / 256, 1, NREL), 256, 0, stream>>>(Wrel0, wrel0F, DH, DH);
    pack_wfrag<<<dim3((DH * DH + 255) / 256, 1, 1), 256, 0, stream>>>(Wloop0, wloop0F, DH, DH);
    pack_wfrag<<<dim3((DH * DH + 255) / 256, 1, NREL), 256, 0, stream>>>(Wrel1, wrel1F, DH, DH);
    pack_wfrag<<<dim3((DH * DH + 255) / 256, 1, 1), 256, 0, stream>>>(Wloop1, wloop1F, DH, DH);

    // Embedding GEMM + ReLU -> X0 (bf16).
    k_embed<<<dim3(8, (BATCH * NNODE) / 128, 1), 256, 0, stream>>>(attn, wembF, bemb, Xrow0);

    const int xfragBlocks = (BATCH * NNODE * DH) / 256;

    // Layer 0.
    pack_xfrag<<<xfragBlocks, 256, 0, stream>>>(Xrow0, Xfrag);
    k_agg<<<dim3(8, 2, BATCH * NREL), 256, 0, stream>>>(rel, Xfrag, relfts);
    k_combine<<<dim3(8, 2, BATCH), 256, 0, stream>>>(relfts, wrel0F, Xrow0, wloop0F, b0,
                                                     Xrow1, nullptr);

    // Layer 1 (final: write f32 to d_out).
    pack_xfrag<<<xfragBlocks, 256, 0, stream>>>(Xrow1, Xfrag);
    k_agg<<<dim3(8, 2, BATCH * NREL), 256, 0, stream>>>(rel, Xfrag, relfts);
    k_combine<<<dim3(8, 2, BATCH), 256, 0, stream>>>(relfts, wrel1F, Xrow1, wloop1F, b1,
                                                     nullptr, out);
}
